// InteractionNet_53506702574084
// MI455X (gfx1250) — compile-verified, matrix operand order checked by reading, not run on hardware
//
#include <hip/hip_runtime.h>
#include <cstdint>
#include <cstddef>

// Problem constants (match reference)
#define FDIM    64
#define GSEG    4096
#define NNODES  1000000
#define FOUT    256
#define KIN     192          // 3*FDIM
#define XSTR    68           // LDS row stride (floats) to spread banks
#define CHROWS  32           // rows staged per chunk (2 WMMA M-subtiles)
#define NEG_BIG (-3.0e38f)

typedef __attribute__((ext_vector_type(16))) _Float16 v16h;
typedef __attribute__((ext_vector_type(8)))  float    v8f;

// K index for element j of a 16-half WMMA A/B fragment (16x32 f16 layout,
// CDNA5 ISA 7.12.2): VGPR p holds K pairs {2p,2p+1} (p<4) / {16+2(p-4),...}
// for lane-half 0; +8 for lane-half 1.
__device__ __forceinline__ int wmma_k(int j, int half) {
  int p  = j >> 1;
  int lo = j & 1;
  int k  = (p < 4) ? (2 * p + lo) : (16 + 2 * (p - 4) + lo);
  return k + 8 * half;
}

// Async copy 16B global -> LDS (CDNA5 GLOBAL_LOAD_ASYNC_TO_LDS_B128, GVS
// mode: LDS-offset VGPR, 32-bit global byte offset VGPR, 64-bit SGPR base).
// Tracked with ASYNCcnt; data lands in LDS without touching VGPRs.
__device__ __forceinline__ void async_b128_to_lds(uint32_t lds_off,
                                                  int g_off,
                                                  const float* base) {
  asm volatile("global_load_async_to_lds_b128 %0, %1, %2"
               :: "v"(lds_off), "v"(g_off), "s"(base)
               : "memory");
}
__device__ __forceinline__ void wait_asynccnt0() {
  asm volatile("s_wait_asynccnt 0x0" ::: "memory");
}

// ---------------------------------------------------------------------------
// Kernel 1: fused  gate = sigmoid(x @ Wg^T + bg)  +  per-channel segment
// softmax + weighted scatter-sum, using online (rescaled) softmax so each
// x element is read from HBM exactly once. One workgroup per (plane, seg),
// double-buffered async global->LDS staging.
// ---------------------------------------------------------------------------
__global__ __launch_bounds__(128)
void attnpool_wmma_kernel(const float* __restrict__ xu, const float* __restrict__ Wu,
                          const float* __restrict__ bu, const int* __restrict__ iu,
                          const float* __restrict__ xv, const float* __restrict__ Wv,
                          const float* __restrict__ bv, const int* __restrict__ iv,
                          const float* __restrict__ xy, const float* __restrict__ Wy,
                          const float* __restrict__ by, const int* __restrict__ iy,
                          float* __restrict__ pooled)   // [3][GSEG][FDIM]
{
  const int g     = blockIdx.x;   // segment
  const int plane = blockIdx.y;   // 0=u 1=v 2=y

  const float* x   = (plane == 0) ? xu : (plane == 1) ? xv : xy;
  const float* W   = (plane == 0) ? Wu : (plane == 1) ? Wv : Wy;
  const float* bgp = (plane == 0) ? bu : (plane == 1) ? bv : by;
  const int*   idx = (plane == 0) ? iu : (plane == 1) ? iv : iy;

  __shared__ float xs[2][CHROWS * XSTR];   // double-buffered x chunks (f32)
  __shared__ int   sb[2];                  // segment [start, end)

  const int tid = threadIdx.x;

  // idx is sorted: binary search segment bounds.
  if (tid < 2) {
    const int target = g + tid;
    int lo = 0, hi = NNODES;
    while (lo < hi) {
      int mid = (lo + hi) >> 1;
      if (idx[mid] < target) lo = mid + 1; else hi = mid;
    }
    sb[tid] = lo;
  }
  __syncthreads();
  const int s_start = sb[0];
  const int s_end   = sb[1];
  const int seglen  = s_end - s_start;

  const int wave = tid >> 5;        // 4 waves; wave w owns channels [16w,16w+16)
  const int lane = tid & 31;
  const int half = lane >> 4;
  const int lm   = lane & 15;
  const int col  = 16 * wave + lm;  // channel 0..63 this lane tracks

  // Resident B fragments: B[k][n=col] = Wg[col*FDIM + k] (gate = x @ Wg^T).
  v16h bf0, bf1;
  for (int j = 0; j < 16; ++j) {
    const int k = wmma_k(j, half);
    bf0[j] = (_Float16)W[col * FDIM + k];
    bf1[j] = (_Float16)W[col * FDIM + k + 32];
  }
  const float bias = bgp[col];

  // Online softmax state for this lane's channel over its 8-row tile halves.
  float m = NEG_BIG, s = 0.0f, vx = 0.0f;

  // Issue one chunk of async B128 copies: CHROWS rows x 16 float4.
  // Pad rows are clamped in-bounds; their values are masked at consumption.
  auto issue_chunk = [&](int base, int buf) {
    for (int i = tid; i < CHROWS * 16; i += 128) {
      const int r   = i >> 4;
      const int c4  = (i & 15) << 2;
      int row = base + r;
      row = (row < s_end) ? row : (NNODES - 1);     // stay in-bounds
      const uint32_t lds_off =
          (uint32_t)(uintptr_t)(&xs[buf][r * XSTR + c4]);
      const int g_off = (row * FDIM + c4) * 4;      // < 256 MB, fits i32
      async_b128_to_lds(lds_off, g_off, x);
    }
  };

  const int nchunks = (seglen + CHROWS - 1) / CHROWS;
  if (nchunks > 0) issue_chunk(s_start, 0);

  for (int c = 0; c < nchunks; ++c) {
    wait_asynccnt0();          // my chunk-c copies landed in LDS
    __syncthreads();           // everyone's chunk-c copies landed

    if (c + 1 < nchunks) issue_chunk(s_start + (c + 1) * CHROWS, (c + 1) & 1);

    const float* xb  = xs[c & 1];
    const int    cb  = s_start + c * CHROWS;

    for (int sub = 0; sub < 2; ++sub) {            // two 16-row M-subtiles
      const int tbase = cb + 16 * sub;
      const int nval  = min(16, s_end - tbase);    // uniform per block
      if (nval <= 0) break;
      const float* xt = xb + sub * 16 * XSTR;

      // A fragments (rows of x, f16) per ISA 16-bit A layout.
      v16h a0, a1;
      for (int j = 0; j < 16; ++j) {
        const int k = wmma_k(j, half);
        a0[j] = (_Float16)xt[lm * XSTR + k];
        a1[j] = (_Float16)xt[lm * XSTR + k + 32];
      }

      v8f cacc = {};
      cacc = __builtin_amdgcn_wmma_f32_16x16x32_f16(false, a0, false, bf0,
                                                    (short)0, cacc, false, false);
      cacc = __builtin_amdgcn_wmma_f32_16x16x32_f16(false, a1, false, bf1,
                                                    (short)0, cacc, false, false);

      // gate = sigmoid(c + bias); C layout: VGPR r -> row r + 8*half.
      float gate[8];
      float cm = NEG_BIG;
      for (int r = 0; r < 8; ++r) {
        const int row = r + 8 * half;
        const float t  = cacc[r] + bias;
        const float gv = __builtin_amdgcn_rcpf(1.0f + __expf(-t));
        gate[r] = (row < nval) ? gv : NEG_BIG;
        cm = fmaxf(cm, gate[r]);
      }

      // Online rescale + accumulate  s = sum e,  vx = sum e * x[row,col].
      const float mn    = fmaxf(m, cm);
      const float scale = __expf(m - mn);   // m==NEG_BIG -> 0, no NaN
      s  *= scale;
      vx *= scale;
      for (int r = 0; r < 8; ++r) {
        const int row = r + 8 * half;
        if (row < nval) {
          const float e = __expf(gate[r] - mn);
          s  += e;
          vx += e * xt[row * XSTR + col];
        }
      }
      m = mn;
    }
    __syncthreads();   // all waves done reading buffer (c&1) before reuse
  }

  // Merge the two 8-row halves of each channel (lanes L and L+16).
  const float m2 = __shfl_xor(m, 16);
  const float s2 = __shfl_xor(s, 16);
  const float v2 = __shfl_xor(vx, 16);
  const float mm = fmaxf(m, m2);
  const float e1 = __expf(m - mm);
  const float e2 = __expf(m2 - mm);
  const float st = s * e1 + s2 * e2;
  const float vt = vx * e1 + v2 * e2;

  if (half == 0) {
    const float outv = (st > 0.0f) ? vt * __builtin_amdgcn_rcpf(st) : 0.0f;
    pooled[(size_t)plane * GSEG * FDIM + (size_t)g * FDIM + col] = outv;
  }
}

// ---------------------------------------------------------------------------
// Kernel 2: out[g,o] = concat(pooled)[g,:] @ W_out^T + b_out via WMMA.
// 4096 16x16 tiles (256 g-tiles x 16 o-tiles), one tile per wave, K=192.
// ---------------------------------------------------------------------------
__global__ __launch_bounds__(128)
void head_gemm_wmma_kernel(const float* __restrict__ pooled,  // [3][GSEG][FDIM]
                           const float* __restrict__ Wout,    // [FOUT][KIN]
                           const float* __restrict__ bout,    // [FOUT]
                           float* __restrict__ out)           // [GSEG][FOUT]
{
  const int tid  = threadIdx.x;
  const int wave = tid >> 5;
  const int lane = tid & 31;
  const int half = lane >> 4;
  const int lm   = lane & 15;

  const int t  = blockIdx.x * 4 + wave;  // tile id 0..4095
  const int tm = t >> 4;                 // g-tile 0..255
  const int tn = t & 15;                 // o-tile 0..15
  const int grow = 16 * tm + lm;         // graph row this lane loads for A
  const int ocol = 16 * tn + lm;         // output column this lane loads for B

  v8f c = {};
  for (int ks = 0; ks < 6; ++ks) {
    v16h a, b;
    for (int j = 0; j < 16; ++j) {
      const int k = wmma_k(j, half) + 32 * ks;     // 0..191
      // h[g,k] = pooled[k/64][g][k%64]
      a[j] = (_Float16)pooled[(size_t)(k >> 6) * GSEG * FDIM +
                              (size_t)grow * FDIM + (k & 63)];
      b[j] = (_Float16)Wout[(size_t)ocol * KIN + k];
    }
    c = __builtin_amdgcn_wmma_f32_16x16x32_f16(false, a, false, b, (short)0, c, false, false);
  }

  const float bb = bout[ocol];
  for (int r = 0; r < 8; ++r) {
    const int gg = 16 * tm + r + 8 * half;
    out[(size_t)gg * FOUT + 16 * tn + lm] = c[r] + bb;
  }
}

// ---------------------------------------------------------------------------
// Launch
// ---------------------------------------------------------------------------
extern "C" void kernel_launch(void* const* d_in, const int* in_sizes, int n_in,
                              void* d_out, int out_size, void* d_ws, size_t ws_size,
                              hipStream_t stream) {
  (void)in_sizes; (void)n_in; (void)out_size; (void)ws_size;

  // setup_inputs() order: x_u,Wg_u,bg_u,idx_u, x_v,Wg_v,bg_v,idx_v,
  //                       x_y,Wg_y,bg_y,idx_y, W_out,b_out,n_graphs
  const float* xu = (const float*)d_in[0];
  const float* Wu = (const float*)d_in[1];
  const float* bu = (const float*)d_in[2];
  const int*   iu = (const int*)  d_in[3];
  const float* xv = (const float*)d_in[4];
  const float* Wv = (const float*)d_in[5];
  const float* bv = (const float*)d_in[6];
  const int*   iv = (const int*)  d_in[7];
  const float* xy = (const float*)d_in[8];
  const float* Wy = (const float*)d_in[9];
  const float* by = (const float*)d_in[10];
  const int*   iy = (const int*)  d_in[11];
  const float* Wo = (const float*)d_in[12];
  const float* bo = (const float*)d_in[13];

  float* pooled = (float*)d_ws;          // 3*4096*64 f32 = 3 MB
  float* out    = (float*)d_out;         // [4096][256] f32

  dim3 grid1(GSEG, 3);
  attnpool_wmma_kernel<<<grid1, 128, 0, stream>>>(
      xu, Wu, bu, iu, xv, Wv, bv, iv, xy, Wy, by, iy, pooled);

  head_gemm_wmma_kernel<<<(GSEG / 16) * (FOUT / 16) / 4, 128, 0, stream>>>(
      pooled, Wo, bo, out);
}
